// KANConv2d_34497177322246
// MI455X (gfx1250) — matrix operand order repaired and checked by reading, not compiled, not run on hardware
//
#include <hip/hip_runtime.h>
#include <hip/hip_bf16.h>
#include <cstdint>

typedef __attribute__((ext_vector_type(16))) _Float16 v16h;
typedef __attribute__((ext_vector_type(8)))  float    v8f;

#define CIN   16
#define COUT  64
#define HH    96
#define WW    96
#define LPIX  (HH*WW)      // 9216 pixels per batch image
#define BATCH 8
#define K2    9
#define NG    3
#define NFEAT 1008         // 16*9*7
#define KPAD  1024         // padded K for 32 chunks of 32
#define TM    32           // pixels per workgroup (2 M-tiles of 16)

// ---------------------------------------------------------------------------
// Prep kernel: fold scale_base / scale_spline*coeff into a single f16 weight
// matrix W[KPAD][COUT], stored PRE-SWIZZLED into WMMA B-fragment layout:
//   dword index  ((nt*32 + c)*32 + lane)*8 + v
//   holds packed {W[kb][O], W[kb+1][O]},  kb = c*32 + (lane>=16?8:0) + (v>=4?16:0) + 2(v&3),
//   O = nt*16 + (lane&15)
// Feature order along K:  f = (i*9 + kpos)*7 + t,
//   t=0: silu weight (scale_base), t=1..3: cos g=t-1, t=4..6: sin g=t-4.
// ---------------------------------------------------------------------------
__global__ void kan_prep_weights(const float* __restrict__ scale_base,
                                 const float* __restrict__ scale_spline,
                                 const float* __restrict__ coeff,
                                 uint32_t* __restrict__ wsw) {
  int idx = blockIdx.x * blockDim.x + threadIdx.x;
  if (idx >= 4 * 32 * 32 * 8) return;
  int v    =  idx        & 7;
  int lane = (idx >> 3)  & 31;
  int c    = (idx >> 8)  & 31;
  int nt   =  idx >> 13;
  int O  = nt * 16 + (lane & 15);
  int kb = c * 32 + ((lane >> 4) << 3) + ((v >> 2) << 4) + 2 * (v & 3);
  uint32_t packed = 0;
  #pragma unroll
  for (int h2 = 0; h2 < 2; ++h2) {
    int K = kb + h2;
    float wv = 0.0f;
    if (K < NFEAT) {
      int ik = K / 7, t = K % 7;
      int i = ik / K2, kp = ik % K2;
      if (t == 0) {
        wv = scale_base[(i * COUT + O) * K2 + kp];
      } else {
        int cs = (t <= NG) ? 0 : 1;
        int g  = (t <= NG) ? (t - 1) : (t - 1 - NG);
        float ss = scale_spline[(i * COUT + O) * K2 + kp];
        float cf = coeff[(((i * 2 + cs) * COUT + O) * K2 + kp) * NG + g];
        wv = ss * cf;
      }
    }
    union { _Float16 f; uint16_t u; } cv;
    cv.f = (_Float16)wv;
    packed |= ((uint32_t)cv.u) << (16 * h2);
  }
  wsw[idx] = packed;
}

// XOR-swizzle of 16-byte LDS chunks so all-lanes-same-chunk A-fragment loads
// fan out across banks (row base banks are identical otherwise).
__device__ __forceinline__ int feat_addr(int pix, int f) {
  return pix * KPAD + (((((f >> 3) ^ (pix & 7))) << 3) | (f & 7));
}

// ---------------------------------------------------------------------------
// Main kernel: fused feature construction (LDS, f16) + WMMA GEMM.
// 256 threads = 8 waves. Block owns TM=32 pixels x all 64 output channels.
// Wave w: M-tile (w>>2), N-tile (w&3).  K loop: 32 chunks of 32 (f16 WMMA),
// fully unrolled.
// ---------------------------------------------------------------------------
__global__ __launch_bounds__(256) void kan_conv_wmma(
    const float* __restrict__ x,
    const uint32_t* __restrict__ wsw,
    const float* __restrict__ bias,
    float* __restrict__ out) {
  __shared__ _Float16 feat[TM * KPAD];   // 64 KB -> up to 5 blocks per WGP

  int tid   = threadIdx.x;
  int gpix0 = blockIdx.x * TM;           // 73728 / 32 = 2304 blocks
  int b   = gpix0 / LPIX;                // uniform per block (32 | 9216)
  int hw0 = gpix0 % LPIX;

  int wave = tid >> 5, lane = tid & 31;
  int mt = wave >> 2, nt = wave & 3;
  int ncol  = lane & 15;                 // A row / B-C column within tile
  int khalf = lane >> 4;
  const float4* bbase =
      reinterpret_cast<const float4*>(wsw) + ((size_t)(nt * 32) * 32 + lane) * 2;

  // Warm the weight stream into the WGP-level caches (locality=3 -> near
  // scope; SYS-scope would stop at GL2) while the wave is busy in the
  // transcendental feature stage below.
  #pragma unroll
  for (int c = 0; c < 32; ++c)
    __builtin_prefetch(bbase + (size_t)c * 64, 0, 3);

  // ---- stage features: one sincos per patch scalar + angle-addition ----
  for (int task = tid; task < TM * 144; task += 256) {
    int pix = task / 144;
    int ik  = task % 144;
    int i  = ik / K2, kk = ik % K2;
    int di = kk / 3 - 1, dj = kk % 3 - 1;
    int hw = hw0 + pix;
    int h  = hw / WW, w = hw % WW;
    int hi = h + di, wi = w + dj;
    float p = 0.0f;
    if ((unsigned)hi < (unsigned)HH && (unsigned)wi < (unsigned)WW)
      p = x[((b * CIN + i) * HH + hi) * WW + wi];
    // silu = p * rcp(1 + exp(-p)) : single v_rcp_f32 instead of IEEE divide
    float sil = p * __builtin_amdgcn_rcpf(1.0f + __expf(-p));
    float s1 = __sinf(p), c1 = __cosf(p);
    float c2 = c1 * c1 - s1 * s1, s2 = 2.0f * s1 * c1;
    float c3 = c2 * c1 - s2 * s1, s3 = s2 * c1 + c2 * s1;
    int f0 = ik * 7;
    feat[feat_addr(pix, f0 + 0)] = (_Float16)sil;
    feat[feat_addr(pix, f0 + 1)] = (_Float16)c1;
    feat[feat_addr(pix, f0 + 2)] = (_Float16)c2;
    feat[feat_addr(pix, f0 + 3)] = (_Float16)c3;
    feat[feat_addr(pix, f0 + 4)] = (_Float16)s1;
    feat[feat_addr(pix, f0 + 5)] = (_Float16)s2;
    feat[feat_addr(pix, f0 + 6)] = (_Float16)s3;
  }
  // zero the K padding (1008..1023)
  for (int z = tid; z < TM * (KPAD - NFEAT); z += 256) {
    feat[feat_addr(z / 16, NFEAT + (z % 16))] = (_Float16)0.0f;
  }
  __syncthreads();

  float bv = bias[nt * 16 + ncol];
  v8f acc = {bv, bv, bv, bv, bv, bv, bv, bv};

  int arow = mt * 16 + ncol;             // LDS pixel row for A fragment
  int koff = khalf * 8;                  // ISA 16-bit A layout K offset

  #pragma unroll
  for (int c = 0; c < 32; ++c) {
    union { float4 f[2]; v16h h; } ua, ub;
    ua.f[0] = *reinterpret_cast<const float4*>(feat + feat_addr(arow, c * 32 + koff));
    ua.f[1] = *reinterpret_cast<const float4*>(feat + feat_addr(arow, c * 32 + koff + 16));
    const float4* bp = bbase + (size_t)c * 64;     // 32 lanes * 2 float4 per chunk
    ub.f[0] = bp[0];
    ub.f[1] = bp[1];
    acc = __builtin_amdgcn_wmma_f32_16x16x32_f16(
        /*neg_a=*/false, ua.h, /*neg_b=*/false, ub.h,
        /*c_mod=*/(short)0, acc, /*reuse_a=*/false, /*reuse_b=*/false);
  }

  // C layout: VGPR v, lanes 0-15 -> M=v, lanes 16-31 -> M=v+8; N = lane&15
  int o = nt * 16 + ncol;
  float* optr = out + ((size_t)(b * COUT + o)) * LPIX + hw0 + mt * 16 + khalf * 8;
  float4 o0 = {acc[0], acc[1], acc[2], acc[3]};
  float4 o1 = {acc[4], acc[5], acc[6], acc[7]};
  *reinterpret_cast<float4*>(optr)     = o0;
  *reinterpret_cast<float4*>(optr + 4) = o1;
}

extern "C" void kernel_launch(void* const* d_in, const int* in_sizes, int n_in,
                              void* d_out, int out_size, void* d_ws, size_t ws_size,
                              hipStream_t stream) {
  const float* x            = (const float*)d_in[0];
  const float* scale_base   = (const float*)d_in[1];
  const float* scale_spline = (const float*)d_in[2];
  const float* coeff        = (const float*)d_in[3];
  const float* bias         = (const float*)d_in[4];
  uint32_t* wsw = (uint32_t*)d_ws;   // 128 KB pre-swizzled f16 weights

  kan_prep_weights<<<(4 * 32 * 32 * 8 + 255) / 256, 256, 0, stream>>>(
      scale_base, scale_spline, coeff, wsw);

  int nblocks = (BATCH * LPIX) / TM;   // 2304
  kan_conv_wmma<<<nblocks, 256, 0, stream>>>(x, wsw, bias, (float*)d_out);
}
